// FlipyFlopy_77300821393661
// MI455X (gfx1250) — matrix operand order repaired and checked by reading, not compiled, not run on hardware
//
#include <hip/hip_runtime.h>

typedef __bf16 bf16;
typedef __attribute__((ext_vector_type(16))) __bf16 v16bf;
typedef __attribute__((ext_vector_type(8)))  __bf16 v8bf;
typedef __attribute__((ext_vector_type(8)))  float  v8f;
typedef __attribute__((ext_vector_type(4)))  unsigned int u32x4;
typedef __attribute__((ext_vector_type(8)))  int i32x8;
typedef __attribute__((ext_vector_type(4)))  int i32x4;

#define BSZ   8
#define CHN   512
#define SEQL  512
#define DKD   64
#define DVD   64
#define NH    8
#define UNITS 2048
#define EPSV  1e-5f

#if defined(__has_builtin)
#if __has_builtin(__builtin_amdgcn_tensor_load_to_lds) && \
    __has_builtin(__builtin_amdgcn_s_wait_tensorcnt)
#define USE_TDM 1
#endif
#endif
#ifndef USE_TDM
#define USE_TDM 0
#endif

#if USE_TDM
// ---- Tensor DMA Descriptor builders (CDNA5 ISA ch. 8.3/8.4) -----------------
// group0: [1:0]=count=1, [63:32]=lds_addr, [120:64]=global_addr, [127:126]=type=2
__device__ __forceinline__ u32x4 tdm_group0(unsigned ldsOff, unsigned long long gaddr)
{
    u32x4 g;
    g.x = 1u;
    g.y = ldsOff;
    g.z = (unsigned)(gaddr & 0xFFFFFFFFull);
    g.w = (unsigned)((gaddr >> 32) & 0x01FFFFFFull) | (2u << 30);
    return g;
}
// group1: data_size=1 (2B), pad_enable, pad_interval/amount, 2D tensor + tile dims
__device__ __forceinline__ i32x8 tdm_group1(unsigned td0, unsigned td1,
                                            unsigned tile0, unsigned tile1,
                                            unsigned long long stride0,
                                            unsigned padInt, unsigned padAmt)
{
    i32x8 g;
    g[0] = (int)((1u << 16) | (1u << 20) | (padInt << 22) | (padAmt << 25));
    g[1] = (int)((td0 & 0xFFFFu) << 16);                       // tensor_dim0[15:0] @ bits63:48
    g[2] = (int)((td0 >> 16) | ((td1 & 0xFFFFu) << 16));       // td0 hi | td1 lo
    g[3] = (int)((td1 >> 16) | (tile0 << 16));                 // td1 hi | tile_dim0
    g[4] = (int)tile1;                                         // tile_dim1 (tile_dim2=0)
    g[5] = (int)(stride0 & 0xFFFFFFFFull);                     // tensor_dim0_stride lo
    g[6] = (int)((stride0 >> 32) & 0xFFFFull);                 // stride0 hi (stride1=0)
    g[7] = 0;
    return g;
}

__device__ __forceinline__ void tdm_load(u32x4 g0, i32x8 g1)
{
    i32x4 z4 = {};
#if __clang_major__ >= 23
    i32x8 z8 = {};
    __builtin_amdgcn_tensor_load_to_lds(g0, g1, z4, z4, z8, 0);
#else
    __builtin_amdgcn_tensor_load_to_lds(g0, g1, z4, z4, 0);
#endif
}
#endif  // USE_TDM

// ---------------------------------------------------------------------------
// Generic tiled bf16 GEMM with fused epilogues.
// C = A(MxK) * B(KxN), f32 accumulation via v_wmma_f32_16x16x32_bf16.
// Block: 128 threads (4 waves), 128x64 C tile; each wave owns 32x64
// (2 A-frags x 4 B-frags = 8 WMMA per 32-wide k-step). blockIdx.z = batch.
// MODE: 0=f32 store, 1=bf16 store, 2=QKV scatter, 3=O head-interleave,
//       4=+transposed-input residual (f32), 5=ReLU->bf16, 6=+aux f32 skip-add
// ---------------------------------------------------------------------------
template<int MODE>
__global__ __launch_bounds__(128)
void gemm_bf16_kernel(const bf16* __restrict__ A, long long aStride,
                      const bf16* __restrict__ B, long long bStride,
                      void* __restrict__ O0, void* __restrict__ O1,
                      void* __restrict__ O2, const void* __restrict__ AUX,
                      int M, int N, int K)
{
    __shared__ __attribute__((aligned(16))) bf16 As[128][48]; // 128x32 tile, 16 DW row + 8 DW pad
    __shared__ __attribute__((aligned(16))) bf16 Bs[32][80];  // 32x64 tile, 32 DW row + 8 DW pad

    const int z    = blockIdx.z;
    const bf16* Ab = A + (long long)z * aStride;
    const bf16* Bb = B + (long long)z * bStride;
    const int m0   = blockIdx.y * 128;
    const int n0   = blockIdx.x * 64;
    const int t    = threadIdx.x;
    const int w    = t >> 5;            // wave id 0..3
    const int lane = t & 31;

    v8f acc[2][4] = {};

    // A fragment coords (ISA 7.12.2: 16-bit A 16x32 layout)
    const int arow = (w << 5) + (lane & 15);
    const int koff = (lane >> 4) << 3;   // lanes 0-15: K{0-7,16-23}; 16-31: K{8-15,24-31}

#if USE_TDM
    const unsigned asOff = (unsigned)(unsigned long long)(uintptr_t)&As[0][0];
    const unsigned bsOff = (unsigned)(unsigned long long)(uintptr_t)&Bs[0][0];
    // A: rows of K elems, tile 32x128; LDS pad: 16 DW interval(code 3), 8 DW amount(code 7)
    const i32x8 g1A = tdm_group1((unsigned)K, (unsigned)M, 32u, 128u,
                                 (unsigned long long)K, 3u, 7u);
    // B: rows of N elems, tile 64x32; LDS pad: 32 DW interval(code 4), 8 DW amount(code 7)
    const i32x8 g1B = tdm_group1((unsigned)N, (unsigned)K, 64u, 32u,
                                 (unsigned long long)N, 4u, 7u);
#else
    const int bk = t >> 2;              // 0..31
    const int bn = (t & 3) << 4;        // 0,16,32,48
#endif

    for (int k0 = 0; k0 < K; k0 += 32) {
#if USE_TDM
        if (w == 0) {   // one wave drives the Tensor Data Mover for both tiles
            tdm_load(tdm_group0(asOff,
                (unsigned long long)(uintptr_t)(Ab + (long long)m0 * K + k0)), g1A);
            tdm_load(tdm_group0(bsOff,
                (unsigned long long)(uintptr_t)(Bb + (long long)k0 * N + n0)), g1B);
            __builtin_amdgcn_s_wait_tensorcnt(0);
        }
        __syncthreads();
#else
        {   // fallback: per-thread global->LDS staging
            const bf16* ga = Ab + (long long)(m0 + t) * K + k0;     // thread t: A row t
            *(v8bf*)&As[t][0]  = *(const v8bf*)(ga);
            *(v8bf*)&As[t][8]  = *(const v8bf*)(ga + 8);
            *(v8bf*)&As[t][16] = *(const v8bf*)(ga + 16);
            *(v8bf*)&As[t][24] = *(const v8bf*)(ga + 24);
            const bf16* gb = Bb + (long long)(k0 + bk) * N + (n0 + bn);
            *(v8bf*)&Bs[bk][bn]     = *(const v8bf*)(gb);
            *(v8bf*)&Bs[bk][bn + 8] = *(const v8bf*)(gb + 8);
            __syncthreads();
            if (k0 + 32 < K) {   // global_prefetch_b8 for the next tiles
                __builtin_prefetch(ga + 32, 0, 1);
                __builtin_prefetch(gb + (long long)32 * N, 0, 1);
            }
        }
#endif
        v16bf a[2];
#pragma unroll
        for (int i = 0; i < 2; ++i) {
            v8bf alo = *(const v8bf*)&As[arow + (i << 4)][koff];
            v8bf ahi = *(const v8bf*)&As[arow + (i << 4)][16 + koff];
            a[i] = __builtin_shufflevector(alo, ahi,
                       0,1,2,3,4,5,6,7,8,9,10,11,12,13,14,15);
        }
#pragma unroll
        for (int j = 0; j < 4; ++j) {
            v16bf b;
#pragma unroll
            for (int s = 0; s < 16; ++s)
                b[s] = Bs[(s < 8) ? (koff + s) : (16 + koff + s - 8)][(j << 4) + (lane & 15)];
#pragma unroll
            for (int i = 0; i < 2; ++i)
                acc[i][j] = __builtin_amdgcn_wmma_f32_16x16x32_bf16(
                                false, a[i], false, b, (short)0, acc[i][j], false, false);
        }
        __syncthreads();
    }

    // C/D layout: VGPR r -> M=r (lanes 0-15) / M=r+8 (lanes 16-31), N=lane&15
    const int nlo = lane & 15;
    const int mwv = m0 + (w << 5) + ((lane >> 4) << 3);
#pragma unroll
    for (int i = 0; i < 2; ++i) {
#pragma unroll
        for (int j = 0; j < 4; ++j) {
            const int n = n0 + (j << 4) + nlo;
#pragma unroll
            for (int r = 0; r < 8; ++r) {
                const int m = mwv + (i << 4) + r;
                float val   = acc[i][j][r];
                if (MODE == 0) {
                    ((float*)O0)[(long long)z * M * N + (long long)m * N + n] = val;
                } else if (MODE == 1) {
                    ((bf16*)O0)[(long long)z * M * N + (long long)m * N + n] = (bf16)val;
                } else if (MODE == 2) {      // QKV scatter; z=batch, n in [0,1536)
                    if (n < 512) {           // Q[b][h][s][d]
                        int h = n >> 6, d = n & 63;
                        ((bf16*)O0)[(((long long)(z * NH + h)) * SEQL + m) * DKD + d] = (bf16)val;
                    } else if (n < 1024) {   // K[b][h][d][s]
                        int nn = n - 512; int h = nn >> 6, d = nn & 63;
                        ((bf16*)O1)[(((long long)(z * NH + h)) * DKD + d) * SEQL + m] = (bf16)val;
                    } else {                 // V[b][h][s][d]
                        int nn = n - 1024; int h = nn >> 6, d = nn & 63;
                        ((bf16*)O2)[(((long long)(z * NH + h)) * SEQL + m) * DVD + d] = (bf16)val;
                    }
                } else if (MODE == 3) {      // O[b][q][v*NH+h]; z = b*NH+h, n = v
                    int b = z >> 3, h = z & 7;
                    ((bf16*)O0)[((long long)(b * SEQL + m)) * (DVD * NH) + n * NH + h] = (bf16)val;
                } else if (MODE == 4) {      // + inp[b][n][m] (transposed residual)
                    float add = ((const float*)AUX)[((long long)(z * CHN + n)) * SEQL + m];
                    ((float*)O0)[(long long)z * M * N + (long long)m * N + n] = val + add;
                } else if (MODE == 5) {      // ReLU -> bf16
                    ((bf16*)O0)[(long long)z * M * N + (long long)m * N + n] =
                        (bf16)fmaxf(val, 0.0f);
                } else if (MODE == 6) {      // + aux (same layout)
                    long long idx = (long long)z * M * N + (long long)m * N + n;
                    ((float*)O0)[idx] = val + ((const float*)AUX)[idx];
                }
            }
        }
    }
}

// ---------------------------------------------------------------------------
// Elementwise / reduction kernels
// ---------------------------------------------------------------------------
__device__ __forceinline__ float dsigmoid(float x) { return 1.0f / (1.0f + __expf(-x)); }

// inp (b,c,s) f32 -> X (b,s,c) bf16
__global__ void k_convert_x(const float* __restrict__ inp, bf16* __restrict__ out)
{
    int tid = blockIdx.x * 256 + threadIdx.x;           // 2M
    int c = tid & 511, s = (tid >> 9) & 511, b = tid >> 18;
    out[tid] = (bf16)inp[(((long long)b << 9) | c) * SEQL + s];
}

// Wq/Wk/Wv (c,d,h) f32 -> fused (c, [Q|K|V] h*64+d) bf16, scaled by sigmoid(alpha)
__global__ void k_convert_wqkv(const float* __restrict__ Wq, const float* __restrict__ Wk,
                               const float* __restrict__ Wv, const float* __restrict__ aq,
                               const float* __restrict__ ak, const float* __restrict__ av,
                               bf16* __restrict__ out)
{
    int tid = blockIdx.x * 256 + threadIdx.x;           // 512*1536
    int n = tid % 1536, c = tid / 1536;
    float v;
    if (n < 512)       { int h = n >> 6, d = n & 63;
                         v = dsigmoid(aq[0]) * Wq[(c * DKD + d) * NH + h]; }
    else if (n < 1024) { int nn = n - 512; int h = nn >> 6, d = nn & 63;
                         v = dsigmoid(ak[0]) * Wk[(c * DKD + d) * NH + h]; }
    else               { int nn = n - 1024; int h = nn >> 6, d = nn & 63;
                         v = dsigmoid(av[0]) * Wv[(c * DVD + d) * NH + h]; }
    out[tid] = (bf16)v;
}

__global__ void k_f32_to_bf16(const float* __restrict__ src, bf16* __restrict__ dst, int n)
{
    int tid = blockIdx.x * 256 + threadIdx.x;
    if (tid < n) dst[tid] = (bf16)src[tid];
}

// 8x8 head mix over J[b][h][q][k] (in place f32, or to bf16). mask optional.
__global__ void k_headmix(float* __restrict__ J, const float* __restrict__ Wm,
                          const float* __restrict__ mask, bf16* __restrict__ outbf, int toBf)
{
    __shared__ float Ws[64];
    if (threadIdx.x < 64) Ws[threadIdx.x] = Wm[threadIdx.x];
    __syncthreads();
    long long tid = (long long)blockIdx.x * 256 + threadIdx.x;   // 2M (b,q,k)
    int k = (int)(tid & 511), q = (int)((tid >> 9) & 511), b = (int)(tid >> 18);
    long long base = (long long)b * NH * 262144LL + (long long)q * 512 + k;
    float in[8];
#pragma unroll
    for (int h = 0; h < 8; ++h) in[h] = J[base + (long long)h * 262144LL];
    float mval = mask ? mask[b * SEQL + k] : 0.0f;
#pragma unroll
    for (int g = 0; g < 8; ++g) {
        float acc = -mval;
#pragma unroll
        for (int h = 0; h < 8; ++h) acc += in[h] * Ws[h * 8 + g];
        if (toBf) outbf[base + (long long)g * 262144LL] = (bf16)acc;
        else      J[base + (long long)g * 262144LL] = acc;
    }
}

// softmax over contiguous rows of 512; one wave32 per row
__global__ void k_softmax(float* __restrict__ X)
{
    int wid  = (blockIdx.x * 256 + threadIdx.x) >> 5;   // 32768 rows
    int lane = threadIdx.x & 31;
    float* row = X + (long long)wid * 512;
    float v[16];
    float mx = -1e30f;
#pragma unroll
    for (int i = 0; i < 16; ++i) { v[i] = row[i * 32 + lane]; mx = fmaxf(mx, v[i]); }
#pragma unroll
    for (int o = 16; o > 0; o >>= 1) mx = fmaxf(mx, __shfl_xor(mx, o, 32));
    float s = 0.0f;
#pragma unroll
    for (int i = 0; i < 16; ++i) { v[i] = __expf(v[i] - mx); s += v[i]; }
#pragma unroll
    for (int o = 16; o > 0; o >>= 1) s += __shfl_xor(s, o, 32);
    float inv = 1.0f / s;
#pragma unroll
    for (int i = 0; i < 16; ++i) row[i * 32 + lane] = v[i] * inv;
}

// per-channel stats over X[b][s][c] (rows = BSZ*SEQL, channel = last dim)
__global__ void k_bn_stats(const float* __restrict__ X, float* __restrict__ mean,
                           float* __restrict__ rstd)
{
    const int c = blockIdx.x, t = threadIdx.x, rows = BSZ * SEQL;
    float s = 0.0f, s2 = 0.0f;
    for (int i = t; i < rows; i += 256) {
        float x = X[(long long)i * CHN + c];
        s += x; s2 += x * x;
    }
    __shared__ float sh[256], sh2[256];
    sh[t] = s; sh2[t] = s2; __syncthreads();
    for (int o = 128; o > 0; o >>= 1) {
        if (t < o) { sh[t] += sh[t + o]; sh2[t] += sh2[t + o]; }
        __syncthreads();
    }
    if (t == 0) {
        float m = sh[0] / rows;
        mean[c] = m;
        rstd[c] = rsqrtf(sh2[0] / rows - m * m + EPSV);
    }
}

// BN apply; keeps f32 in place and emits bf16 copy for the FFN GEMM
__global__ void k_bn_apply1(float* __restrict__ X, bf16* __restrict__ Xbf,
                            const float* __restrict__ mean, const float* __restrict__ rstd,
                            const float* __restrict__ gamma, const float* __restrict__ beta)
{
    int tid = blockIdx.x * 256 + threadIdx.x;           // 2M
    int c = tid & 511;
    float y = (X[tid] - mean[c]) * rstd[c] * gamma[c] + beta[c];
    X[tid] = y;
    Xbf[tid] = (bf16)y;
}

// BN apply + transpose (b,s,c) -> out (b,c,s)
__global__ void k_bn_apply2(const float* __restrict__ X, float* __restrict__ out,
                            const float* __restrict__ mean, const float* __restrict__ rstd,
                            const float* __restrict__ gamma, const float* __restrict__ beta)
{
    int tid = blockIdx.x * 256 + threadIdx.x;           // 2M
    int c = tid & 511, s = (tid >> 9) & 511, b = tid >> 18;
    float y = (X[tid] - mean[c]) * rstd[c] * gamma[c] + beta[c];
    out[((long long)(b * CHN + c)) * SEQL + s] = y;
}

// ---------------------------------------------------------------------------
// Workspace layout (bytes)
// ---------------------------------------------------------------------------
static const size_t OFF_XBF   = 0;                        // 4,194,304  bf16 X (b,s,c)
static const size_t OFF_WQKV  = 4194304;                  // 1,572,864
static const size_t OFF_WO    = 5767168;                  //   524,288
static const size_t OFF_W1    = 6291456;                  // 2,097,152
static const size_t OFF_W2    = 8388608;                  // 2,097,152
static const size_t OFF_QBF   = 10485760;                 // 4,194,304
static const size_t OFF_KBF   = 14680064;                 // 4,194,304
static const size_t OFF_VBF   = 18874368;                 // 4,194,304
static const size_t OFF_JBUF  = 23068672;                 // 67,108,864 f32 J/EL (in-place)
static const size_t OFF_UBF   = 90177536;                 // 33,554,432 bf16 U
static const size_t OFF_STATS = 123731968;                // 8,192 (mean/rstd x2)
// aliases inside JBUF (free after post-mix writes UBF):
static const size_t OFF_OBF   = OFF_JBUF + 0;             // 4,194,304  bf16
static const size_t OFF_X1    = OFF_JBUF + 4194304;       // 8,388,608  f32
static const size_t OFF_X1BF  = OFF_JBUF + 12582912;      // 4,194,304  bf16
static const size_t OFF_R1BF  = OFF_JBUF + 16777216;      // 16,777,216 bf16
static const size_t OFF_X2    = OFF_JBUF + 33554432;      // 8,388,608  f32

extern "C" void kernel_launch(void* const* d_in, const int* in_sizes, int n_in,
                              void* d_out, int out_size, void* d_ws, size_t ws_size,
                              hipStream_t stream)
{
    (void)in_sizes; (void)n_in; (void)out_size; (void)ws_size;
    const float* inp    = (const float*)d_in[0];
    const float* mask   = (const float*)d_in[1];
    const float* Wq     = (const float*)d_in[2];
    const float* Wk     = (const float*)d_in[3];
    const float* Wv     = (const float*)d_in[4];
    const float* alphaq = (const float*)d_in[5];
    const float* alphak = (const float*)d_in[6];
    const float* alphav = (const float*)d_in[7];
    const float* Wl     = (const float*)d_in[8];
    const float* Ww     = (const float*)d_in[9];
    const float* Wo     = (const float*)d_in[10];
    const float* gamma1 = (const float*)d_in[11];
    const float* beta1  = (const float*)d_in[12];
    const float* W1     = (const float*)d_in[13];
    const float* W2     = (const float*)d_in[14];
    const float* gamma2 = (const float*)d_in[15];
    const float* beta2  = (const float*)d_in[16];

    char* ws = (char*)d_ws;
    bf16*  XBF   = (bf16*)(ws + OFF_XBF);
    bf16*  WQKV  = (bf16*)(ws + OFF_WQKV);
    bf16*  WOB   = (bf16*)(ws + OFF_WO);
    bf16*  W1B   = (bf16*)(ws + OFF_W1);
    bf16*  W2B   = (bf16*)(ws + OFF_W2);
    bf16*  QBF   = (bf16*)(ws + OFF_QBF);
    bf16*  KBF   = (bf16*)(ws + OFF_KBF);
    bf16*  VBF   = (bf16*)(ws + OFF_VBF);
    float* JBUF  = (float*)(ws + OFF_JBUF);
    bf16*  UBF   = (bf16*)(ws + OFF_UBF);
    bf16*  OBF   = (bf16*)(ws + OFF_OBF);
    float* X1    = (float*)(ws + OFF_X1);
    bf16*  X1BF  = (bf16*)(ws + OFF_X1BF);
    bf16*  R1BF  = (bf16*)(ws + OFF_R1BF);
    float* X2    = (float*)(ws + OFF_X2);
    float* MEAN1 = (float*)(ws + OFF_STATS);
    float* RSTD1 = MEAN1 + 512;
    float* MEAN2 = MEAN1 + 1024;
    float* RSTD2 = MEAN1 + 1536;

    // --- precision conversion / weight fusion ---
    k_convert_x   <<<8192, 256, 0, stream>>>(inp, XBF);
    k_convert_wqkv<<<3072, 256, 0, stream>>>(Wq, Wk, Wv, alphaq, alphak, alphav, WQKV);
    k_f32_to_bf16 <<<1024, 256, 0, stream>>>(Wo, WOB, 262144);
    k_f32_to_bf16 <<<4096, 256, 0, stream>>>(W1, W1B, 1048576);
    k_f32_to_bf16 <<<4096, 256, 0, stream>>>(W2, W2B, 1048576);

    // --- QKV projection: (b) [512 x 1536 x 512], scatter to attention layouts ---
    gemm_bf16_kernel<2><<<dim3(24, 4, 8), 128, 0, stream>>>(
        XBF, 262144LL, WQKV, 0LL, QBF, KBF, VBF, nullptr, 512, 1536, 512);

    // --- J = Q K^T : (b,h) [512 x 512 x 64] ---
    gemm_bf16_kernel<0><<<dim3(8, 4, 64), 128, 0, stream>>>(
        QBF, 32768LL, KBF, 32768LL, JBUF, nullptr, nullptr, nullptr, 512, 512, 64);

    // --- pre-softmax head mix (Wl) - mask, in place; softmax; post mix (Ww) -> bf16 ---
    k_headmix<<<8192, 256, 0, stream>>>(JBUF, Wl, mask, nullptr, 0);
    k_softmax<<<4096, 256, 0, stream>>>(JBUF);
    k_headmix<<<8192, 256, 0, stream>>>(JBUF, Ww, nullptr, UBF, 1);

    // --- O = U V : (b,h) [512 x 64 x 512], interleave heads into (b,s,512) ---
    gemm_bf16_kernel<3><<<dim3(1, 4, 64), 128, 0, stream>>>(
        UBF, 262144LL, VBF, 32768LL, OBF, nullptr, nullptr, nullptr, 512, 64, 512);

    // --- resid = O Wo + inp^T : (b) [512 x 512 x 512] -> X1 (b,s,c) f32 ---
    gemm_bf16_kernel<4><<<dim3(8, 4, 8), 128, 0, stream>>>(
        OBF, 262144LL, WOB, 0LL, X1, nullptr, nullptr, inp, 512, 512, 512);

    // --- BatchNorm 1 ---
    k_bn_stats <<<512, 256, 0, stream>>>(X1, MEAN1, RSTD1);
    k_bn_apply1<<<8192, 256, 0, stream>>>(X1, X1BF, MEAN1, RSTD1, gamma1, beta1);

    // --- FFN1: relu(X1 W1) : (b) [512 x 2048 x 512] -> bf16 ---
    gemm_bf16_kernel<5><<<dim3(32, 4, 8), 128, 0, stream>>>(
        X1BF, 262144LL, W1B, 0LL, R1BF, nullptr, nullptr, nullptr, 512, 2048, 512);

    // --- FFN2: R1 W2 + X1 : (b) [512 x 512 x 2048] -> f32 ---
    gemm_bf16_kernel<6><<<dim3(8, 4, 8), 128, 0, stream>>>(
        R1BF, 1048576LL, W2B, 0LL, X2, nullptr, nullptr, X1, 512, 512, 2048);

    // --- BatchNorm 2 + transpose to output (b,c,s) ---
    k_bn_stats <<<512, 256, 0, stream>>>(X2, MEAN2, RSTD2);
    k_bn_apply2<<<8192, 256, 0, stream>>>(X2, (float*)d_out, MEAN2, RSTD2, gamma2, beta2);
}
